// KMeansEmbed_60610578481733
// MI455X (gfx1250) — matrix operand order repaired and checked by reading, not compiled, not run on hardware
//
#include <hip/hip_runtime.h>
#include <hip/hip_bf16.h>

// ---------------------------------------------------------------------------
// KMeansEmbed for MI455X (gfx1250): f16 WMMA GEMMs, encoded-atomic segment-max,
// async global->LDS gather for the concat tile.
// ---------------------------------------------------------------------------

#define B_   16
#define N_   8192
#define G_   256
#define BN_  (B_ * N_)

typedef __attribute__((ext_vector_type(16))) _Float16 v16h;
typedef __attribute__((ext_vector_type(8)))  _Float16 v8h;
typedef __attribute__((ext_vector_type(8)))  float    v8f;

__device__ __forceinline__ v16h mk16(v8h lo, v8h hi) {
    return __builtin_shufflevector(lo, hi, 0,1,2,3,4,5,6,7,8,9,10,11,12,13,14,15);
}

// order-preserving float -> uint encoding for atomic segment-max
__device__ __forceinline__ unsigned encf(float f) {
    unsigned u = __float_as_uint(f);
    return (u & 0x80000000u) ? ~u : (u | 0x80000000u);
}
__device__ __forceinline__ float decf(unsigned e) {
    unsigned u = (e & 0x80000000u) ? (e & 0x7FFFFFFFu) : ~e;
    return __uint_as_float(u);
}
__device__ __forceinline__ bool finitef(float f) {
    return (__float_as_uint(f) & 0x7F800000u) != 0x7F800000u;
}
#define ENC_NEG_INF 0x007FFFFFu   // encf(-inf)

// ---------------------------------------------------------------------------
// K0: init accumulators.  Grid = B*G*256 threads exactly (1,048,576).
// ---------------------------------------------------------------------------
__global__ void k_init(float* cent_sum, float* counts,
                       unsigned* pooled_enc, unsigned* out_enc) {
    int idx = blockIdx.x * blockDim.x + threadIdx.x;
    pooled_enc[idx] = ENC_NEG_INF;
    out_enc[idx]    = ENC_NEG_INF;
    if (idx < B_ * G_ * 3) cent_sum[idx] = 0.f;
    if (idx < B_ * G_)     counts[idx]   = 0.f;
}

// ---------------------------------------------------------------------------
// K1: transpose + f16-convert weights.  W_t is [N][K] row-major so B-matrix
// fragments load with the same addressing as A fragments.
// Grid = 512*512 threads (262,144).
// ---------------------------------------------------------------------------
__global__ void k_prepw(const float* w1b, const float* w2a, const float* w2b,
                        _Float16* w1b_t, _Float16* w2a_t, _Float16* w2b_t) {
    int idx = blockIdx.x * blockDim.x + threadIdx.x;
    { // w2a [512][512] -> w2a_t[n*512+k]
        int n = idx & 511, k = idx >> 9;
        w2a_t[n * 512 + k] = (_Float16)w2a[k * 512 + n];
    }
    if (idx < 128 * 256) { // w1b [128][256] -> w1b_t[n*128+k]
        int n = idx & 255, k = idx >> 8;
        w1b_t[n * 128 + k] = (_Float16)w1b[k * 256 + n];
    }
    if (idx < 512 * 256) { // w2b [512][256] -> w2b_t[n*512+k]
        int n = idx & 255, k = idx >> 8;
        w2b_t[n * 512 + k] = (_Float16)w2b[k * 256 + n];
    }
}

// ---------------------------------------------------------------------------
// K2: centroid accumulation (atomic).  Grid = BN_ threads.
// ---------------------------------------------------------------------------
__global__ void k_cent_accum(const float* xyz, const int* labels,
                             float* cent_sum, float* counts) {
    int row = blockIdx.x * blockDim.x + threadIdx.x;
    int b = row >> 13;                 // / N_
    int g = labels[row];
    int base = (b << 8) + g;
    atomicAdd(&cent_sum[base * 3 + 0], xyz[row * 3 + 0]);
    atomicAdd(&cent_sum[base * 3 + 1], xyz[row * 3 + 1]);
    atomicAdd(&cent_sum[base * 3 + 2], xyz[row * 3 + 2]);
    atomicAdd(&counts[base], 1.f);
}

// K3: finalize centroids -> output section 0.  Grid = B*G threads.
__global__ void k_cent_fin(const float* cent_sum, const float* counts,
                           float* out_cent) {
    int idx = blockIdx.x * blockDim.x + threadIdx.x;
    float c = fmaxf(counts[idx], 1.f);
    out_cent[idx * 3 + 0] = cent_sum[idx * 3 + 0] / c;
    out_cent[idx * 3 + 1] = cent_sum[idx * 3 + 1] / c;
    out_cent[idx * 3 + 2] = cent_sum[idx * 3 + 2] / c;
}

// ---------------------------------------------------------------------------
// K4: MLP1 fused.  64 points/block, 256 threads (8 wave32).
//   phase A: rel_xyz, Linear(3,128)+LN+ReLU -> f16 tile in LDS (stride 136)
//            also emits p_i and labels outputs
//   phase B: WMMA GEMM [64x128] x [128x256] -> h; store h f16, atomicMax pool
// ---------------------------------------------------------------------------
__global__ __launch_bounds__(256)
void k_mlp1(const float* xyz, const int* labels, const float* cent,
            const float* w1a, const float* b1a, const float* g1, const float* beta1,
            const _Float16* w1b_t, const float* b1b,
            _Float16* h_f16, unsigned* pooled_enc,
            float* out_pi, float* out_lab) {
    __shared__ float    s_w1a[384];
    __shared__ float    s_b1a[128], s_g1[128], s_beta1[128];
    __shared__ _Float16 s_A[64 * 136];      // padded stride: no bank conflicts
    __shared__ int      s_lab[64];

    const int tid = threadIdx.x;
    const int r0  = blockIdx.x * 64;

    for (int i = tid; i < 384; i += 256) s_w1a[i] = w1a[i];
    if (tid < 128) { s_b1a[tid] = b1a[tid]; s_g1[tid] = g1[tid]; s_beta1[tid] = beta1[tid]; }

    const int p = tid >> 2, sub = tid & 3;
    const int row = r0 + p;
    const int b = row >> 13;
    const int g = labels[row];
    if (sub == 0) { s_lab[p] = g; out_lab[row] = (float)g; }
    const float* cp = cent + ((b << 8) + g) * 3;
    float c0 = cp[0], c1 = cp[1], c2 = cp[2];
    if (sub == 0) { out_pi[row*3] = c0; out_pi[row*3+1] = c1; out_pi[row*3+2] = c2; }
    const float rx = xyz[row*3] - c0, ry = xyz[row*3+1] - c1, rz = xyz[row*3+2] - c2;
    __syncthreads();

    float a[32], sum = 0.f, ss = 0.f;
#pragma unroll
    for (int jj = 0; jj < 32; jj++) {
        int j = (sub << 5) + jj;
        float v = s_b1a[j] + rx * s_w1a[j] + ry * s_w1a[128 + j] + rz * s_w1a[256 + j];
        a[jj] = v; sum += v; ss += v * v;
    }
    sum += __shfl_xor(sum, 1); sum += __shfl_xor(sum, 2);
    ss  += __shfl_xor(ss, 1);  ss  += __shfl_xor(ss, 2);
    float mean = sum * (1.f / 128.f);
    float var  = ss * (1.f / 128.f) - mean * mean;
    float rstd = rsqrtf(var + 1e-5f);
#pragma unroll
    for (int jj = 0; jj < 32; jj++) {
        int j = (sub << 5) + jj;
        float v = (a[jj] - mean) * rstd * s_g1[j] + s_beta1[j];
        s_A[p * 136 + j] = (_Float16)fmaxf(v, 0.f);
    }
    __syncthreads();

    // WMMA: wave -> (mtile, 128-col half)
    const int wave = tid >> 5, lane = tid & 31;
    const int mtile = wave & 3, nbase = (wave >> 2) * 128;
    v8f acc[8] = {};
#pragma unroll
    for (int k = 0; k < 4; k++) {
        // A frag (16-bit 16x32 ISA layout): lane holds row lane%16,
        // K = (lane>=16?8:0)+{0..7} and the same +16.
        const _Float16* ap = &s_A[(mtile*16 + (lane & 15)) * 136 + k*32 + ((lane >> 4) << 3)];
        v16h af = mk16(*(const v8h*)ap, *(const v8h*)(ap + 16));
#pragma unroll
        for (int nt = 0; nt < 8; nt++) {
            int n = nbase + nt*16 + (lane & 15);
            const _Float16* bp = w1b_t + n*128 + k*32 + ((lane >> 4) << 3);
            v16h bf = mk16(*(const v8h*)bp, *(const v8h*)(bp + 16));
            acc[nt] = __builtin_amdgcn_wmma_f32_16x16x32_f16(
                false, af, false, bf, (short)0, acc[nt], false, false);
        }
    }
#pragma unroll
    for (int nt = 0; nt < 8; nt++) {
        int n = nbase + nt*16 + (lane & 15);
        float bias = b1b[n];
#pragma unroll
        for (int i = 0; i < 8; i++) {
            int m = ((lane >> 4) << 3) + i;
            int prow = r0 + mtile*16 + m;
            float x = acc[nt][i] + bias;
            h_f16[(prow << 8) + n] = (_Float16)x;
            int gg = s_lab[mtile*16 + m];
            int bb = prow >> 13;
            atomicMax(&pooled_enc[(((bb << 8) + gg) << 8) + n], encf(x));
        }
    }
}

// K5: decode pooled max (non-finite -> 0) to f16.  Grid = B*G*256.
__global__ void k_pool_decode(const unsigned* pooled_enc, _Float16* pooled_f16) {
    int idx = blockIdx.x * blockDim.x + threadIdx.x;
    float f = decf(pooled_enc[idx]);
    if (!finitef(f)) f = 0.f;
    pooled_f16[idx] = (_Float16)f;
}

// ---------------------------------------------------------------------------
// K6: MLP2 fully fused. 64 points/block, 256 threads, ~195 KB dynamic LDS.
//   A: async gather concat [64x512] f16 directly into LDS (ASYNCcnt path)
//   B: WMMA [64x512]x[512x512] -> f32 scores in LDS (+b2a)
//   C: LN(512)+ReLU -> f16 back into the A buffer
//   D: WMMA [64x512]x[512x256] (+b2b) -> encoded atomicMax segment max
// ---------------------------------------------------------------------------
#define AS2 520   // f16 row stride (halves)
#define DS2 520   // f32 row stride (floats)
#define SMEM_A_BYTES  (64 * AS2 * 2)                 // 66560
#define SMEM_D_BYTES  (64 * DS2 * 4)                 // 133120
#define SMEM_LAB_OFF  (SMEM_A_BYTES + SMEM_D_BYTES)  // 199680
#define SMEM_TOTAL    (SMEM_LAB_OFF + 256)           // 199936

__global__ __launch_bounds__(256)
void k_mlp2(const int* labels, const _Float16* pooled_f16, const _Float16* h_f16,
            const _Float16* w2a_t, const float* b2a, const float* g2, const float* beta2,
            const _Float16* w2b_t, const float* b2b, unsigned* out_enc) {
    extern __shared__ char smem[];
    _Float16* sA  = (_Float16*)smem;
    float*    sD  = (float*)(smem + SMEM_A_BYTES);
    int*      sLab = (int*)(smem + SMEM_LAB_OFF);

    const int tid = threadIdx.x;
    const int r0  = blockIdx.x * 64;
    if (tid < 64) sLab[tid] = labels[r0 + tid];
    __syncthreads();

    // gather concat: cat = [pooled[label], h]  (512 f16 per point)
    // GLOBAL_LOAD_ASYNC_TO_LDS_B128: 16B per lane, global -> LDS directly,
    // tracked by ASYNCcnt (no VGPR round-trip, no DScnt pressure).
    // This kernel has only dynamic LDS, so LDS byte offsets start at 0.
    for (int c = tid; c < 4096; c += 256) {       // 4096 chunks of 8 halves
        int p = c >> 6;
        int off = (c & 63) << 3;
        int row = r0 + p, b = row >> 13;
        const _Float16* src = (off < 256)
            ? pooled_f16 + (((b << 8) + sLab[p]) << 8) + off
            : h_f16 + (row << 8) + (off - 256);
        unsigned           lds_off = (unsigned)(p * AS2 + off) * 2u;   // bytes
        unsigned long long gaddr   = (unsigned long long)(uintptr_t)src;
        asm volatile("global_load_async_to_lds_b128 %0, %1, off"
                     :: "v"(lds_off), "v"(gaddr) : "memory");
    }
    asm volatile("s_wait_asynccnt 0x0" ::: "memory");
    __syncthreads();

    const int wave = tid >> 5, lane = tid & 31;

    { // GEMM 2a: 64x512x512, each wave: 1 mtile x 256 cols (16 ntiles)
        const int mtile = wave & 3, nbase = (wave >> 2) * 256;
        v8f acc[16] = {};
#pragma unroll 1
        for (int k = 0; k < 16; k++) {
            const _Float16* ap = &sA[(mtile*16 + (lane & 15)) * AS2 + k*32 + ((lane >> 4) << 3)];
            v16h af = mk16(*(const v8h*)ap, *(const v8h*)(ap + 16));
#pragma unroll
            for (int nt = 0; nt < 16; nt++) {
                int n = nbase + nt*16 + (lane & 15);
                const _Float16* bp = w2a_t + (n << 9) + k*32 + ((lane >> 4) << 3);
                __builtin_prefetch(bp + 32, 0, 0);   // global_prefetch next k-line
                v16h bf = mk16(*(const v8h*)bp, *(const v8h*)(bp + 16));
                acc[nt] = __builtin_amdgcn_wmma_f32_16x16x32_f16(
                    false, af, false, bf, (short)0, acc[nt], false, false);
            }
        }
#pragma unroll
        for (int nt = 0; nt < 16; nt++) {
            int n = nbase + nt*16 + (lane & 15);
            float bias = b2a[n];
#pragma unroll
            for (int i = 0; i < 8; i++) {
                int m = ((lane >> 4) << 3) + i;
                sD[(mtile*16 + m) * DS2 + n] = acc[nt][i] + bias;
            }
        }
    }
    __syncthreads();

    { // LayerNorm(512) + ReLU -> f16, overwrite A buffer
        const int p = tid >> 2, sub = tid & 3;
        float sum = 0.f, ss = 0.f;
        for (int jj = 0; jj < 128; jj++) {
            float v = sD[p * DS2 + (sub << 7) + jj];
            sum += v; ss += v * v;
        }
        sum += __shfl_xor(sum, 1); sum += __shfl_xor(sum, 2);
        ss  += __shfl_xor(ss, 1);  ss  += __shfl_xor(ss, 2);
        float mean = sum * (1.f / 512.f);
        float var  = ss * (1.f / 512.f) - mean * mean;
        float rstd = rsqrtf(var + 1e-5f);
        for (int jj = 0; jj < 128; jj++) {
            int j = (sub << 7) + jj;
            float v = (sD[p * DS2 + j] - mean) * rstd * g2[j] + beta2[j];
            sA[p * AS2 + j] = (_Float16)fmaxf(v, 0.f);
        }
    }
    __syncthreads();

    { // GEMM 2b: 64x512x256, each wave: 1 mtile x 128 cols (8 ntiles)
        const int mtile = wave & 3, nbase = (wave >> 2) * 128;
        v8f acc[8] = {};
#pragma unroll 1
        for (int k = 0; k < 16; k++) {
            const _Float16* ap = &sA[(mtile*16 + (lane & 15)) * AS2 + k*32 + ((lane >> 4) << 3)];
            v16h af = mk16(*(const v8h*)ap, *(const v8h*)(ap + 16));
#pragma unroll
            for (int nt = 0; nt < 8; nt++) {
                int n = nbase + nt*16 + (lane & 15);
                const _Float16* bp = w2b_t + (n << 9) + k*32 + ((lane >> 4) << 3);
                v16h bf = mk16(*(const v8h*)bp, *(const v8h*)(bp + 16));
                acc[nt] = __builtin_amdgcn_wmma_f32_16x16x32_f16(
                    false, af, false, bf, (short)0, acc[nt], false, false);
            }
        }
#pragma unroll
        for (int nt = 0; nt < 8; nt++) {
            int n = nbase + nt*16 + (lane & 15);
            float bias = b2b[n];
#pragma unroll
            for (int i = 0; i < 8; i++) {
                int m = ((lane >> 4) << 3) + i;
                int prow = r0 + mtile*16 + m;
                int bb = prow >> 13;
                int gg = sLab[mtile*16 + m];
                float x = acc[nt][i] + bias;
                atomicMax(&out_enc[(((bb << 8) + gg) << 8) + n], encf(x));
            }
        }
    }
}

// K7: decode final segment max -> output section 1.  Grid = B*G*256.
__global__ void k_out_decode(const unsigned* out_enc, float* out) {
    int idx = blockIdx.x * blockDim.x + threadIdx.x;
    float f = decf(out_enc[idx]);
    out[idx] = finitef(f) ? f : 0.f;
}

// ---------------------------------------------------------------------------
// launch
// ---------------------------------------------------------------------------
extern "C" void kernel_launch(void* const* d_in, const int* in_sizes, int n_in,
                              void* d_out, int out_size, void* d_ws, size_t ws_size,
                              hipStream_t stream) {
    (void)in_sizes; (void)n_in; (void)out_size; (void)ws_size;

    const float* xyz    = (const float*)d_in[0];
    const int*   labels = (const int*)  d_in[1];
    const float* w1a    = (const float*)d_in[2];
    const float* b1a    = (const float*)d_in[3];
    const float* g1     = (const float*)d_in[4];
    const float* beta1  = (const float*)d_in[5];
    const float* w1b    = (const float*)d_in[6];
    const float* b1b    = (const float*)d_in[7];
    const float* w2a    = (const float*)d_in[8];
    const float* b2a    = (const float*)d_in[9];
    const float* g2     = (const float*)d_in[10];
    const float* beta2  = (const float*)d_in[11];
    const float* w2b    = (const float*)d_in[12];
    const float* b2b    = (const float*)d_in[13];

    // output layout (floats): centroids | out | p_i | labels
    float* o        = (float*)d_out;
    float* out_cent = o;                       // B*G*3    = 12288
    float* out_seg  = o + 12288;               // B*G*256  = 1048576
    float* out_pi   = o + 1060864;             // B*N*3    = 393216
    float* out_lab  = o + 1454080;             // B*N      = 131072

    // workspace layout (bytes)
    char* ws = (char*)d_ws;
    float*     cent_sum   = (float*)    (ws + 0);          //  49152 B
    float*     counts     = (float*)    (ws + 49152);      //  16384 B
    _Float16*  w1b_t      = (_Float16*) (ws + 65536);      //  65536 B
    _Float16*  w2a_t      = (_Float16*) (ws + 131072);     // 524288 B
    _Float16*  w2b_t      = (_Float16*) (ws + 655360);     // 262144 B
    unsigned*  pooled_enc = (unsigned*) (ws + 917504);     // 4 MiB
    _Float16*  pooled_f16 = (_Float16*) (ws + 5111808);    // 2 MiB
    unsigned*  out_enc    = (unsigned*) (ws + 7208960);    // 4 MiB
    _Float16*  h_f16      = (_Float16*) (ws + 11403264);   // 64 MiB

    const dim3 blk(256);

    k_init      <<<dim3(4096), blk, 0, stream>>>(cent_sum, counts, pooled_enc, out_enc);
    k_prepw     <<<dim3(1024), blk, 0, stream>>>(w1b, w2a, w2b, w1b_t, w2a_t, w2b_t);
    k_cent_accum<<<dim3(512),  blk, 0, stream>>>(xyz, labels, cent_sum, counts);
    k_cent_fin  <<<dim3(16),   blk, 0, stream>>>(cent_sum, counts, out_cent);
    k_mlp1      <<<dim3(2048), blk, 0, stream>>>(xyz, labels, out_cent,
                                                 w1a, b1a, g1, beta1, w1b_t, b1b,
                                                 h_f16, pooled_enc, out_pi, out_lab);
    k_pool_decode<<<dim3(4096), blk, 0, stream>>>(pooled_enc, pooled_f16);
    k_mlp2      <<<dim3(2048), blk, SMEM_TOTAL, stream>>>(labels, pooled_f16, h_f16,
                                                          w2a_t, b2a, g2, beta2,
                                                          w2b_t, b2b, out_enc);
    k_out_decode<<<dim3(4096), blk, 0, stream>>>(out_enc, out_seg);
}
